// DifferentialCrossAttentionLayer_8615704396333
// MI455X (gfx1250) — compile-verified, matrix operand order checked by reading, not run on hardware
//
#include <hip/hip_runtime.h>
#include <hip/hip_bf16.h>
#include <math.h>

// ---------------------------------------------------------------------------
// Types for CDNA5 WMMA (wave32)
// ---------------------------------------------------------------------------
typedef __attribute__((ext_vector_type(16))) __bf16 v16bf;
typedef __attribute__((ext_vector_type(8)))  __bf16 v8bf;
typedef __attribute__((ext_vector_type(8)))  float  v8f;

#define DEV __device__ __forceinline__

static constexpr int D_MODEL = 1024;
static constexpr int NUM_HEADS = 8;
static constexpr int D_HEAD = 128;
static constexpr int D_FF = 4096;
static constexpr int BATCH = 2;
static constexpr int SEQ = 2048;       // Sq == Sk
static constexpr int MTOT = BATCH * SEQ; // 4096 rows for projections

DEV v8f wmma_bf16f32(v16bf a, v16bf b, v8f c) {
  // D = A(16x32 bf16) * B(32x16 bf16) + C(16x16 f32)
  return __builtin_amdgcn_wmma_f32_16x16x32_bf16(
      /*neg_a=*/false, a, /*neg_b=*/false, b,
      /*c_mod=*/(short)0, c, /*reuse_a=*/false, /*reuse_b=*/false);
}

// Load a 16x32 bf16 fragment (A-layout, ISA 7.12.2) from a row-major source.
// Lane L holds row m = L%16; K chunks: [c0..c0+7] and [16+c0..16+c0+7],
// c0 = (L/16)*8.  Two contiguous 16-byte loads per lane.
// Used for A fragments and (assuming lane<->column symmetry) B fragments
// loaded from a [N,K] row-major (i.e. pre-transposed) operand.
DEV v16bf load_frag(const __bf16* __restrict__ p, int ld) {
  const int lane = (int)(threadIdx.x & 31);
  const __bf16* r = p + (size_t)(lane & 15) * (size_t)ld + ((lane >> 4) << 3);
  v8bf lo = *(const v8bf*)(r);
  v8bf hi = *(const v8bf*)(r + 16);
  v16bf f;
#pragma unroll
  for (int i = 0; i < 8; ++i) { f[i] = lo[i]; f[i + 8] = hi[i]; }
  return f;
}

// ---------------------------------------------------------------------------
// Elementwise f32 -> bf16
// ---------------------------------------------------------------------------
__global__ __launch_bounds__(256) void f32_to_bf16_kernel(
    const float* __restrict__ in, __bf16* __restrict__ out, int n) {
  int i = blockIdx.x * 256 + threadIdx.x;
  if (i < n) out[i] = (__bf16)in[i];
}

// ---------------------------------------------------------------------------
// Transpose + convert: in f32 [R,C] row-major -> out bf16 [C,R] row-major
// ---------------------------------------------------------------------------
__global__ __launch_bounds__(256) void transpose_f32_to_bf16_kernel(
    const float* __restrict__ in, __bf16* __restrict__ out, int R, int C) {
  __shared__ float tile[32][33];
  int c0 = blockIdx.x * 32, r0 = blockIdx.y * 32;
  int tx = threadIdx.x, ty = threadIdx.y; // block (32,8)
#pragma unroll
  for (int i = 0; i < 4; ++i)
    tile[ty + 8 * i][tx] = in[(size_t)(r0 + ty + 8 * i) * C + (c0 + tx)];
  __syncthreads();
#pragma unroll
  for (int i = 0; i < 4; ++i)
    out[(size_t)(c0 + ty + 8 * i) * R + (r0 + tx)] = (__bf16)tile[tx][ty + 8 * i];
}

// ---------------------------------------------------------------------------
// WMMA GEMM:  C[M,N] = A[M,K](bf16, row-major) * Bt[N,K](bf16, row-major)^T
// Block: 128x128, 8 waves; wave tile 32x64 -> 8 v8f accumulators,
// 8 v_wmma per K-step of 32.
// OUT_MODE: 0 = f32, 1 = bf16, 2 = bf16 transposed per-head V layout
//           (out[b][h][d][sk], for the attention P*V operand)
// ---------------------------------------------------------------------------
template <int OUT_MODE, bool BIAS, bool RELU>
__global__ __launch_bounds__(256) void gemm_wmma_bf16(
    const __bf16* __restrict__ A, const __bf16* __restrict__ Bt,
    void* __restrict__ Cout, const float* __restrict__ bias,
    int M, int N, int K) {
  const int w = (int)(threadIdx.x >> 5);
  const int lane = (int)(threadIdx.x & 31);
  const int wm = w & 3;       // 4 waves along M
  const int wn = w >> 2;      // 2 waves along N
  const int m0 = blockIdx.y * 128 + wm * 32;
  const int n0 = blockIdx.x * 128 + wn * 64;

  v8f acc[2][4];
#pragma unroll
  for (int i = 0; i < 2; ++i)
#pragma unroll
    for (int t = 0; t < 4; ++t) {
      v8f z = {0.f, 0.f, 0.f, 0.f, 0.f, 0.f, 0.f, 0.f};
      acc[i][t] = z;
    }

  const __bf16* Arow0 = A + (size_t)m0 * K;
  const __bf16* Arow1 = A + (size_t)(m0 + 16) * K;

  for (int k = 0; k < K; k += 32) {
    if (k + 32 < K) { // pull next K-slice toward L2/L0 (global_prefetch_b8)
      __builtin_prefetch((const void*)(Arow0 + (size_t)(lane & 15) * K + k + 32), 0, 1);
      __builtin_prefetch((const void*)(Bt + (size_t)(n0 + (lane & 15)) * K + k + 32), 0, 1);
    }
    v16bf a0 = load_frag(Arow0 + k, K);
    v16bf a1 = load_frag(Arow1 + k, K);
#pragma unroll
    for (int t = 0; t < 4; ++t) {
      v16bf b = load_frag(Bt + (size_t)(n0 + 16 * t) * K + k, K);
      acc[0][t] = wmma_bf16f32(a0, b, acc[0][t]);
      acc[1][t] = wmma_bf16f32(a1, b, acc[1][t]);
    }
  }

  // Epilogue. C-tile layout: VGPR j of lane -> row m = j + 8*(lane>=16),
  // col n = lane & 15 (ISA 7.12.2).
  const int half = lane >> 4, nl = lane & 15;
#pragma unroll
  for (int mi = 0; mi < 2; ++mi)
#pragma unroll
    for (int t = 0; t < 4; ++t)
#pragma unroll
      for (int j = 0; j < 8; ++j) {
        float v = acc[mi][t][j];
        int r = m0 + 16 * mi + j + 8 * half;
        int c = n0 + 16 * t + nl;
        if (BIAS) v += bias[c];
        if (RELU) v = fmaxf(v, 0.f);
        if (OUT_MODE == 0) {
          ((float*)Cout)[(size_t)r * N + c] = v;
        } else if (OUT_MODE == 1) {
          ((__bf16*)Cout)[(size_t)r * N + c] = (__bf16)v;
        } else {
          // V projection written as Vt[b][h][d][sk] for contiguous
          // B-fragments in the attention P*V WMMA.
          int bb = r >> 11, sk = r & 2047;     // r over B*Sk
          int hh = c >> 7, dd = c & 127;       // c over H*Dh
          size_t idx = ((size_t)((bb * NUM_HEADS + hh) * D_HEAD + dd)) * SEQ + sk;
          ((__bf16*)Cout)[idx] = (__bf16)v;
        }
      }
}

// ---------------------------------------------------------------------------
// Flash differential attention.
// softmax(S-1) == softmax(S)  =>  attn = (1 - sigmoid(lambda)) * softmax(S).
// Grid: (Sq/128, H, B); 8 waves/block; each wave owns 16 q-rows x Dh=128.
// Per 32-key step: 8 WMMA (S = Q*K^T) + online softmax + 8 WMMA (O += P*V).
// P round-trips through per-wave LDS (DS ops are in-order within a wave).
// ---------------------------------------------------------------------------
__global__ __launch_bounds__(256) void flash_diff_attn_kernel(
    const __bf16* __restrict__ Q,   // [B,Sq,D_MODEL]
    const __bf16* __restrict__ Kp,  // [B,Sk,D_MODEL]
    const __bf16* __restrict__ Vt,  // [B,H,Dh,Sk]
    const float* __restrict__ lambda_p,
    __bf16* __restrict__ Aout) {    // [B,Sq,D_MODEL]
  const int b = blockIdx.z, h = blockIdx.y;
  const int w = (int)(threadIdx.x >> 5);
  const int lane = (int)(threadIdx.x & 31);
  const int half = lane >> 4, nl = lane & 15;
  const int q0 = blockIdx.x * 128 + w * 16;

  __shared__ __align__(16) __bf16 Pbuf[8][16 * 32];
  __bf16* P = Pbuf[w];

  const __bf16* Qb = Q + ((size_t)b * SEQ + q0) * D_MODEL + h * D_HEAD;
  const __bf16* Kb = Kp + (size_t)b * SEQ * D_MODEL + h * D_HEAD;
  const __bf16* Vb = Vt + (size_t)(b * NUM_HEADS + h) * D_HEAD * SEQ;

  v16bf qf[4];
#pragma unroll
  for (int kk = 0; kk < 4; ++kk) qf[kk] = load_frag(Qb + kk * 32, D_MODEL);

  v8f O[8];
#pragma unroll
  for (int t = 0; t < 8; ++t) {
    v8f z = {0.f, 0.f, 0.f, 0.f, 0.f, 0.f, 0.f, 0.f};
    O[t] = z;
  }
  float mrow[8], lrow[8];
#pragma unroll
  for (int j = 0; j < 8; ++j) { mrow[j] = -3.0e38f; lrow[j] = 0.f; }

  const float scale = 0.08838834764831845f; // 1/sqrt(128)

  for (int k0 = 0; k0 < SEQ; k0 += 32) {
    v8f s0, s1;
    {
      v8f z = {0.f, 0.f, 0.f, 0.f, 0.f, 0.f, 0.f, 0.f};
      s0 = z; s1 = z;
    }
#pragma unroll
    for (int kk = 0; kk < 4; ++kk) { // K-dim = D_HEAD = 4 * 32
      v16bf kf0 = load_frag(Kb + (size_t)k0 * D_MODEL + kk * 32, D_MODEL);
      v16bf kf1 = load_frag(Kb + (size_t)(k0 + 16) * D_MODEL + kk * 32, D_MODEL);
      s0 = wmma_bf16f32(qf[kk], kf0, s0);
      s1 = wmma_bf16f32(qf[kk], kf1, s1);
    }
    // Online softmax across the 32 keys of this step. Row m = j + 8*half
    // lives across the lane's 16-lane half -> butterfly reductions stay
    // within the half (xor masks < 16).
#pragma unroll
    for (int j = 0; j < 8; ++j) {
      float a0 = s0[j] * scale;
      float a1 = s1[j] * scale;
      float vmax = fmaxf(a0, a1);
#pragma unroll
      for (int m = 1; m < 16; m <<= 1) vmax = fmaxf(vmax, __shfl_xor(vmax, m, 32));
      float mnew = fmaxf(mrow[j], vmax);
      float corr = __expf(mrow[j] - mnew);
      float p0 = __expf(a0 - mnew);
      float p1 = __expf(a1 - mnew);
      float rs = p0 + p1;
#pragma unroll
      for (int m = 1; m < 16; m <<= 1) rs += __shfl_xor(rs, m, 32);
      lrow[j] = lrow[j] * corr + rs;
      mrow[j] = mnew;
#pragma unroll
      for (int t = 0; t < 8; ++t) O[t][j] *= corr;
      int r = j + 8 * half;
      P[r * 32 + nl] = (__bf16)p0;
      P[r * 32 + 16 + nl] = (__bf16)p1;
    }
    // O += P * V   (A = P 16x32 from LDS, B = Vt rows [d, keys])
    v16bf pf = load_frag(P, 32);
#pragma unroll
    for (int t = 0; t < 8; ++t) {
      v16bf vf = load_frag(Vb + (size_t)(16 * t) * SEQ + k0, SEQ);
      O[t] = wmma_bf16f32(pf, vf, O[t]);
    }
  }

  float lam = 1.f / (1.f + __expf(-lambda_p[0]));
  float coef = 1.f - lam; // attn1 - lam*attn2 == (1-lam)*softmax
#pragma unroll
  for (int t = 0; t < 8; ++t)
#pragma unroll
    for (int j = 0; j < 8; ++j) {
      int r = q0 + j + 8 * half;
      int c = h * D_HEAD + 16 * t + nl;
      Aout[((size_t)b * SEQ + r) * D_MODEL + c] = (__bf16)(O[t][j] * coef / lrow[j]);
    }
}

// ---------------------------------------------------------------------------
// out = LayerNorm(X + Y) * gamma + beta ; optional bf16 copy for next GEMM.
// One block (256 threads) per row of 1024.
// ---------------------------------------------------------------------------
template <bool WRITE_BF16>
__global__ __launch_bounds__(256) void residual_layernorm_kernel(
    const float* __restrict__ X, const float* __restrict__ Y,
    const float* __restrict__ gamma, const float* __restrict__ beta,
    float* __restrict__ out_f, __bf16* __restrict__ out_h) {
  const int row = blockIdx.x;
  const int tid = threadIdx.x;
  __shared__ float red[256];
  float v[4];
  float s = 0.f;
#pragma unroll
  for (int i = 0; i < 4; ++i) {
    int c = tid + 256 * i;
    v[i] = X[(size_t)row * D_MODEL + c] + Y[(size_t)row * D_MODEL + c];
    s += v[i];
  }
  red[tid] = s; __syncthreads();
  for (int st = 128; st > 0; st >>= 1) {
    if (tid < st) red[tid] += red[tid + st];
    __syncthreads();
  }
  float mean = red[0] * (1.0f / (float)D_MODEL);
  __syncthreads();
  float vs = 0.f;
#pragma unroll
  for (int i = 0; i < 4; ++i) { float d = v[i] - mean; vs += d * d; }
  red[tid] = vs; __syncthreads();
  for (int st = 128; st > 0; st >>= 1) {
    if (tid < st) red[tid] += red[tid + st];
    __syncthreads();
  }
  float rstd = rsqrtf(red[0] * (1.0f / (float)D_MODEL) + 1e-5f);
#pragma unroll
  for (int i = 0; i < 4; ++i) {
    int c = tid + 256 * i;
    float y = (v[i] - mean) * rstd * gamma[c] + beta[c];
    out_f[(size_t)row * D_MODEL + c] = y;
    if (WRITE_BF16) out_h[(size_t)row * D_MODEL + c] = (__bf16)y;
  }
}

// ---------------------------------------------------------------------------
// Host launcher
// ---------------------------------------------------------------------------
extern "C" void kernel_launch(void* const* d_in, const int* in_sizes, int n_in,
                              void* d_out, int out_size, void* d_ws, size_t ws_size,
                              hipStream_t stream) {
  (void)in_sizes; (void)n_in; (void)out_size; (void)ws_size;
  const float* query = (const float*)d_in[0];
  const float* key_i = (const float*)d_in[1];
  const float* value = (const float*)d_in[2];
  const float* Wq    = (const float*)d_in[3];
  const float* Wk    = (const float*)d_in[4];
  const float* Wv    = (const float*)d_in[5];
  const float* Wo    = (const float*)d_in[6];
  const float* lam   = (const float*)d_in[7];
  const float* ln1g  = (const float*)d_in[8];
  const float* ln1b  = (const float*)d_in[9];
  const float* ln2g  = (const float*)d_in[10];
  const float* ln2b  = (const float*)d_in[11];
  const float* W1    = (const float*)d_in[12];
  const float* b1    = (const float*)d_in[13];
  const float* W2    = (const float*)d_in[14];
  const float* b2    = (const float*)d_in[15];

  // Workspace carving (256B aligned)
  char* p = (char*)d_ws;
  auto alloc = [&](size_t bytes) -> void* {
    void* r = (void*)p;
    p += (bytes + 255) & ~(size_t)255;
    return r;
  };
  const size_t ND = (size_t)MTOT * D_MODEL;   // 4096*1024
  __bf16* qh     = (__bf16*)alloc(ND * 2);
  __bf16* kh     = (__bf16*)alloc(ND * 2);
  __bf16* vh     = (__bf16*)alloc(ND * 2);
  __bf16* WqT    = (__bf16*)alloc((size_t)D_MODEL * D_MODEL * 2);
  __bf16* WkT    = (__bf16*)alloc((size_t)D_MODEL * D_MODEL * 2);
  __bf16* WvT    = (__bf16*)alloc((size_t)D_MODEL * D_MODEL * 2);
  __bf16* WoT    = (__bf16*)alloc((size_t)D_MODEL * D_MODEL * 2);
  __bf16* W1T    = (__bf16*)alloc((size_t)D_MODEL * D_FF * 2);
  __bf16* W2T    = (__bf16*)alloc((size_t)D_FF * D_MODEL * 2);
  __bf16* Qp     = (__bf16*)alloc(ND * 2);
  __bf16* Kp     = (__bf16*)alloc(ND * 2);
  __bf16* Vt     = (__bf16*)alloc(ND * 2);  // [B,H,Dh,Sk]
  __bf16* attn_h = (__bf16*)alloc(ND * 2);
  float*  attnWo = (float*)alloc(ND * 4);
  float*  qenh   = (float*)alloc(ND * 4);
  __bf16* qenh_h = (__bf16*)alloc(ND * 2);
  __bf16* ffn1   = (__bf16*)alloc((size_t)MTOT * D_FF * 2);
  float*  ffn2   = (float*)alloc(ND * 4);

  const int nElem = (int)ND;
  dim3 blk256(256);
  dim3 cvtGrid((nElem + 255) / 256);
  f32_to_bf16_kernel<<<cvtGrid, blk256, 0, stream>>>(query, qh, nElem);
  f32_to_bf16_kernel<<<cvtGrid, blk256, 0, stream>>>(key_i, kh, nElem);
  f32_to_bf16_kernel<<<cvtGrid, blk256, 0, stream>>>(value, vh, nElem);

  dim3 tblk(32, 8);
  transpose_f32_to_bf16_kernel<<<dim3(D_MODEL / 32, D_MODEL / 32), tblk, 0, stream>>>(Wq, WqT, D_MODEL, D_MODEL);
  transpose_f32_to_bf16_kernel<<<dim3(D_MODEL / 32, D_MODEL / 32), tblk, 0, stream>>>(Wk, WkT, D_MODEL, D_MODEL);
  transpose_f32_to_bf16_kernel<<<dim3(D_MODEL / 32, D_MODEL / 32), tblk, 0, stream>>>(Wv, WvT, D_MODEL, D_MODEL);
  transpose_f32_to_bf16_kernel<<<dim3(D_MODEL / 32, D_MODEL / 32), tblk, 0, stream>>>(Wo, WoT, D_MODEL, D_MODEL);
  transpose_f32_to_bf16_kernel<<<dim3(D_FF / 32, D_MODEL / 32), tblk, 0, stream>>>(W1, W1T, D_MODEL, D_FF);
  transpose_f32_to_bf16_kernel<<<dim3(D_MODEL / 32, D_FF / 32), tblk, 0, stream>>>(W2, W2T, D_FF, D_MODEL);

  // Projections
  dim3 gD(D_MODEL / 128, MTOT / 128);            // (8, 32)
  gemm_wmma_bf16<1, false, false><<<gD, blk256, 0, stream>>>(qh, WqT, Qp, nullptr, MTOT, D_MODEL, D_MODEL);
  gemm_wmma_bf16<1, false, false><<<gD, blk256, 0, stream>>>(kh, WkT, Kp, nullptr, MTOT, D_MODEL, D_MODEL);
  gemm_wmma_bf16<2, false, false><<<gD, blk256, 0, stream>>>(vh, WvT, Vt, nullptr, MTOT, D_MODEL, D_MODEL);

  // Flash differential attention
  flash_diff_attn_kernel<<<dim3(SEQ / 128, NUM_HEADS, BATCH), blk256, 0, stream>>>(Qp, Kp, Vt, lam, attn_h);

  // Output projection + residual LN1
  gemm_wmma_bf16<0, false, false><<<gD, blk256, 0, stream>>>(attn_h, WoT, attnWo, nullptr, MTOT, D_MODEL, D_MODEL);
  residual_layernorm_kernel<true><<<dim3(MTOT), blk256, 0, stream>>>(query, attnWo, ln1g, ln1b, qenh, qenh_h);

  // FFN
  dim3 gF(D_FF / 128, MTOT / 128);               // (32, 32)
  gemm_wmma_bf16<1, true, true><<<gF, blk256, 0, stream>>>(qenh_h, W1T, ffn1, b1, MTOT, D_FF, D_MODEL);
  gemm_wmma_bf16<0, true, false><<<gD, blk256, 0, stream>>>(ffn1, W2T, ffn2, b2, MTOT, D_MODEL, D_FF);

  // Residual LN2 -> output
  residual_layernorm_kernel<false><<<dim3(MTOT), blk256, 0, stream>>>(qenh, ffn2, ln2g, ln2b, (float*)d_out, nullptr);
}
